// Net_78262894068350
// MI455X (gfx1250) — matrix operand order repaired and checked by reading, not compiled, not run on hardware
//
#include <hip/hip_runtime.h>
#include <hip/hip_bf16.h>
#include <stdint.h>

#define N_NODES 10000
#define N_EDGES 80000
#define DDIM    1024

#define BM 128
#define BN 128
#define BK 32
#define NKT ((2 * DDIM) / BK)   // 64 k-steps over fused K=2048

typedef __attribute__((ext_vector_type(16))) __bf16          v16bf;
typedef __attribute__((ext_vector_type(8)))  float           v8f;
typedef __attribute__((ext_vector_type(8)))  unsigned short  v8us;
typedef __attribute__((ext_vector_type(16))) unsigned short  v16us;
typedef __attribute__((ext_vector_type(4)))  unsigned short  us4;

union frag_u { v16us u; v16bf b; };

// 16-byte int vector, matching the async-LDS builtin's parameter type
typedef int v4i_t __attribute__((vector_size(16)));
typedef __attribute__((address_space(1))) v4i_t g_v4i;
typedef __attribute__((address_space(3))) v4i_t l_v4i;

__device__ __forceinline__ unsigned short f32_to_bf16_rne(float f) {
    union { float f; unsigned int u; } v; v.f = f;
    unsigned int u = v.u;
    u += 0x7FFFu + ((u >> 16) & 1u);   // round to nearest even
    return (unsigned short)(u >> 16);
}

// ---- CDNA5 async global->LDS (ASYNCcnt-tracked DMA), 16B per lane ----
__device__ __forceinline__ void async_ld16(const unsigned short* g, unsigned short* l) {
#if __has_builtin(__builtin_amdgcn_global_load_async_to_lds_b128)
    __builtin_amdgcn_global_load_async_to_lds_b128(
        (g_v4i*)g, (l_v4i*)l, 0, 0);
#else
    unsigned int laddr = (unsigned int)(size_t)(__attribute__((address_space(3))) void*)l;
    unsigned long long gaddr = (unsigned long long)(size_t)g;
    asm volatile("global_load_async_to_lds_b128 %0, %1, off"
                 :: "v"(laddr), "v"(gaddr) : "memory");
#endif
}

template <int N>
__device__ __forceinline__ void wait_asynccnt() {
#if __has_builtin(__builtin_amdgcn_s_wait_asynccnt)
    __builtin_amdgcn_s_wait_asynccnt(N);
#else
    asm volatile("s_wait_asynccnt %0" :: "i"(N));
#endif
}

// ---------------- scatter: h_neigh[dst] += h[src], 1 block per edge ----------------
__global__ __launch_bounds__(256)
void scatter_add_kernel(const float* __restrict__ h,
                        const int* __restrict__ src,
                        const int* __restrict__ dst,
                        float* __restrict__ out) {
    const int e = blockIdx.x;
    const int s = src[e];
    const int d = dst[e];
    const int t = threadIdx.x * 4;
    const float4 v = *(const float4*)(h + (size_t)s * DDIM + t);
    float* o = out + (size_t)d * DDIM + t;
    unsafeAtomicAdd(o + 0, v.x);
    unsafeAtomicAdd(o + 1, v.y);
    unsafeAtomicAdd(o + 2, v.z);
    unsafeAtomicAdd(o + 3, v.w);
}

// ---------------- fp32 -> bf16 (4 elems / thread) ----------------
__global__ __launch_bounds__(256)
void f32_to_bf16_kernel(const float* __restrict__ in, unsigned short* __restrict__ out, int n) {
    const int i = (blockIdx.x * 256 + threadIdx.x) * 4;
    if (i >= n) return;
    const float4 v = *(const float4*)(in + i);
    us4 o;
    o.x = f32_to_bf16_rne(v.x);
    o.y = f32_to_bf16_rne(v.y);
    o.z = f32_to_bf16_rne(v.z);
    o.w = f32_to_bf16_rne(v.w);
    *(us4*)(out + i) = o;
}

// ---------------- fp32 W[k][n] -> bf16 Wt[n][k] ----------------
__global__ __launch_bounds__(256)
void w_transpose_bf16_kernel(const float* __restrict__ w, unsigned short* __restrict__ wt) {
    const int idx = blockIdx.x * 256 + threadIdx.x;   // 0 .. 1024*1024-1
    const int k = idx >> 10;
    const int n = idx & 1023;
    wt[(size_t)n * DDIM + k] = f32_to_bf16_rne(w[idx]);
}

// ---------------- fused SAGE GEMM with async-LDS double buffering ----------------
// Ab/Hb: [N_NODES][1024] bf16 row-major.  WtS/WtN: [1024(n)][1024(k)] bf16 (pre-transposed).
// Logical K = 2048: k<1024 -> (Ab, WtS), k>=1024 -> (Hb, WtN).
__global__ __launch_bounds__(256)
void sage_gemm_kernel(const unsigned short* __restrict__ Ab,
                      const unsigned short* __restrict__ Hb,
                      const unsigned short* __restrict__ WtS,
                      const unsigned short* __restrict__ WtN,
                      const float* __restrict__ bias,
                      float* __restrict__ out) {
    __shared__ __attribute__((aligned(16))) unsigned short Asm[2][BM * BK];
    __shared__ __attribute__((aligned(16))) unsigned short Bsm[2][BN * BK];

    const int tid  = threadIdx.x;
    const int wave = tid >> 5;          // 0..7, owns 16-row strip
    const int lane = tid & 31;
    const int g    = lane >> 4;         // lane half
    const int ln   = lane & 15;

    const int m0 = blockIdx.x * BM;
    const int n0 = blockIdx.y * BN;

    v8f acc[8];
#pragma unroll
    for (int t = 0; t < 8; ++t)
#pragma unroll
        for (int j = 0; j < 8; ++j) acc[t][j] = 0.0f;

    // Issue one 32-k-deep tile pair (A:128x32, B:128x32) as async DMA into LDS buf.
    // 512 16B chunks per tile; each thread issues 2 chunks per tile (4 asyncs/wave/step).
    auto issue_tile = [&](int kk, int buf) {
        const unsigned short* Aptr = (kk < DDIM) ? Ab  : Hb;
        const unsigned short* Wptr = (kk < DDIM) ? WtS : WtN;
        const int kloc = kk & (DDIM - 1);
#pragma unroll
        for (int q = 0; q < 2; ++q) {
            const int c   = tid * 2 + q;      // chunk 0..511
            const int row = c >> 2;           // tile row 0..127
            const int kc  = (c & 3) * 8;      // ushort offset in 32-wide k slice
            int rowg = m0 + row;              // clamp instead of zero-fill: rows
            if (rowg > N_NODES - 1) rowg = N_NODES - 1;  // >=N_NODES never stored
            async_ld16(Aptr + (size_t)rowg * DDIM + kloc + kc, &Asm[buf][row * BK + kc]);
            async_ld16(Wptr + (size_t)(n0 + row) * DDIM + kloc + kc, &Bsm[buf][row * BK + kc]);
        }
    };

    issue_tile(0, 0);

    for (int i = 0; i < NKT; ++i) {
        if (i + 1 < NKT) {
            issue_tile((i + 1) * BK, (i + 1) & 1);
            wait_asynccnt<4>();               // tile i landed; tile i+1 in flight
        } else {
            wait_asynccnt<0>();
        }
        __syncthreads();

        const unsigned short* As = Asm[i & 1];
        const unsigned short* Bs = Bsm[i & 1];

        // A fragment, ISA 16-bit A 16x32 layout:
        //   lane group g holds K = {g*8..g*8+7, 16+g*8..16+g*8+7} for row ln
        frag_u af;
        {
            const unsigned short* base = As + (wave * 16 + ln) * BK;
            v8us lo = *(const v8us*)(base + g * 8);
            v8us hi = *(const v8us*)(base + 16 + g * 8);
#pragma unroll
            for (int j = 0; j < 8; ++j) { af.u[j] = lo[j]; af.u[8 + j] = hi[j]; }
        }

        // Preload all 8 B fragments, then issue WMMAs back-to-back.
        frag_u bfr[8];
#pragma unroll
        for (int t = 0; t < 8; ++t) {
            const unsigned short* base = Bs + (t * 16 + ln) * BK + g * 8;
            v8us lo = *(const v8us*)(base);
            v8us hi = *(const v8us*)(base + 16);
#pragma unroll
            for (int j = 0; j < 8; ++j) { bfr[t].u[j] = lo[j]; bfr[t].u[8 + j] = hi[j]; }
        }
#pragma unroll
        for (int t = 0; t < 8; ++t) {
            acc[t] = __builtin_amdgcn_wmma_f32_16x16x32_bf16(
                false, af.b, false, bfr[t].b, (short)0, acc[t], false, false);
        }
        __syncthreads();
    }

    // Epilogue: C/D layout — VGPR v: lanes 0-15 -> M=v, lanes 16-31 -> M=8+v; N = ln
#pragma unroll
    for (int t = 0; t < 8; ++t) {
        const int col = n0 + t * 16 + ln;
        const float bv = bias[col];
#pragma unroll
        for (int v = 0; v < 8; ++v) {
            const int row = m0 + wave * 16 + g * 8 + v;
            if (row < N_NODES)
                out[(size_t)row * DDIM + col] = acc[t][v] + bv;
        }
    }
}

extern "C" void kernel_launch(void* const* d_in, const int* in_sizes, int n_in,
                              void* d_out, int out_size, void* d_ws, size_t ws_size,
                              hipStream_t stream) {
    const float* x   = (const float*)d_in[0];
    const int*   ei  = (const int*)d_in[1];
    const float* Ws1 = (const float*)d_in[2];
    const float* Wn1 = (const float*)d_in[3];
    const float* b1  = (const float*)d_in[4];
    const float* Ws2 = (const float*)d_in[5];
    const float* Wn2 = (const float*)d_in[6];
    const float* b2  = (const float*)d_in[7];
    float* out = (float*)d_out;
    const int* src = ei;
    const int* dst = ei + N_EDGES;

    const size_t nodef = (size_t)N_NODES * DDIM;       // 10,240,000
    const size_t wf    = (size_t)DDIM * DDIM;          // 1,048,576

    char* ws = (char*)d_ws;
    float*          hneigh = (float*)ws;          ws += nodef * 4;
    float*          h1     = (float*)ws;          ws += nodef * 4;
    unsigned short* xb     = (unsigned short*)ws; ws += nodef * 2;
    unsigned short* hnb    = (unsigned short*)ws; ws += nodef * 2;
    unsigned short* wtS    = (unsigned short*)ws; ws += wf * 2;
    unsigned short* wtN    = (unsigned short*)ws; ws += wf * 2;

    const dim3 gGemm((N_NODES + BM - 1) / BM, DDIM / BN);
    const int  gCvt  = (int)(nodef / 1024);   // 4 elems/thread, 256 threads
    const int  gWT   = (int)(wf / 256);

    // ---------------- layer 1 ----------------
    (void)hipMemsetAsync(hneigh, 0, nodef * 4, stream);
    scatter_add_kernel<<<N_EDGES, 256, 0, stream>>>(x, src, dst, hneigh);
    f32_to_bf16_kernel<<<gCvt, 256, 0, stream>>>(x, xb, (int)nodef);
    f32_to_bf16_kernel<<<gCvt, 256, 0, stream>>>(hneigh, hnb, (int)nodef);
    w_transpose_bf16_kernel<<<gWT, 256, 0, stream>>>(Ws1, wtS);
    w_transpose_bf16_kernel<<<gWT, 256, 0, stream>>>(Wn1, wtN);
    sage_gemm_kernel<<<gGemm, 256, 0, stream>>>(xb, hnb, wtS, wtN, b1, h1);

    // ---------------- layer 2 ----------------
    (void)hipMemsetAsync(hneigh, 0, nodef * 4, stream);
    scatter_add_kernel<<<N_EDGES, 256, 0, stream>>>(h1, src, dst, hneigh);
    f32_to_bf16_kernel<<<gCvt, 256, 0, stream>>>(h1, xb, (int)nodef);
    f32_to_bf16_kernel<<<gCvt, 256, 0, stream>>>(hneigh, hnb, (int)nodef);
    w_transpose_bf16_kernel<<<gWT, 256, 0, stream>>>(Ws2, wtS);
    w_transpose_bf16_kernel<<<gWT, 256, 0, stream>>>(Wn2, wtN);
    sage_gemm_kernel<<<gGemm, 256, 0, stream>>>(xb, hnb, wtS, wtN, b2, out);
}
